// HybridRegression_65481071410436
// MI455X (gfx1250) — matrix operand order repaired and checked by reading, not compiled, not run on hardware
//
#include <hip/hip_runtime.h>
#include <hip/hip_bf16.h>

typedef __attribute__((ext_vector_type(16))) _Float16 v16h;
typedef __attribute__((ext_vector_type(8)))  float    v8f;

#define NSTATE 65536
#define THREADS 512
#define BSZ 512

// ---------------- GF(2)-linear index maps (CNOT chains), compile-time foldable ----
__device__ __forceinline__ constexpr uint32_t cxg(uint32_t x, int c, int t) {
  return x ^ (((x >> (15 - c)) & 1u) << (15 - t));
}
// random-layer chain CX(10,11),(9,10),(8,9),(7,8),(6,7),(5,6): gather applies last gate first
__device__ __forceinline__ constexpr uint32_t gP(uint32_t x) {
  x = cxg(x, 5, 6); x = cxg(x, 6, 7); x = cxg(x, 7, 8);
  x = cxg(x, 8, 9); x = cxg(x, 9, 10); x = cxg(x, 10, 11);
  return x;
}
// ring chain CX(0,1)..CX(14,15),CX(15,0): gather applies last gate first
__device__ __forceinline__ constexpr uint32_t gQ(uint32_t x) {
  x = cxg(x, 15, 0);
  for (int w = 14; w >= 0; --w) x = cxg(x, w, w + 1);
  return x;
}
// cumulative layout maps: phys = lam(logical)
template<int LAM>
__device__ __forceinline__ constexpr uint32_t lam(uint32_t x) {
  if constexpr (LAM == 1) return gP(x);                 // after random-layer perm
  else if constexpr (LAM == 2) return gP(gQ(x));        // after first ring
  else return gP(gQ(gQ(x)));                            // after second ring
}

struct Meas {
  float aTot; float aBit[16];   // attn1 accumulators (pre-RY1)
  float mTot; float mBit[16];   // m1 accumulators   (post-RY1)
};

// ---------------- one fused 4-wire stage pass: 32x32 real matrix via 2 WMMAs -----
template<int LAM, int PNIB, bool PRE, bool POST, bool STORE>
__device__ void stage_pass(_Float16* __restrict__ Re, _Float16* __restrict__ Im,
                           const _Float16* __restrict__ mat, Meas& ms) {
  const int lane = threadIdx.x & 31;
  const int wave = threadIdx.x >> 5;            // 16 waves
  const int n    = lane & 15;                   // column / base within tile
  const int im   = lane >> 4;                   // 0: Re half (K 0..15), 1: Im half (K 16..31)
  const int K0   = im * 16;

  // A fragments (16x32 f16, contiguous K): lane L: row L&15, K0=(L>=16)*16
  v16h aT, aB;
#pragma unroll
  for (int j = 0; j < 8; ++j) {
    aT[j]     = mat[n * 32 + K0 + j];
    aT[8 + j] = mat[n * 32 + K0 + 8 + j];
    aB[j]     = mat[(n + 16) * 32 + K0 + j];
    aB[8 + j] = mat[(n + 16) * 32 + K0 + 8 + j];
  }
  const _Float16* __restrict__ src = im ? Im : Re;
  const uint32_t imoff = im ? lam<LAM>(8u << (4 * PNIB)) : 0u;

#pragma unroll 1
  for (int it = 0; it < 16; ++it) {
    const uint32_t t = (uint32_t)wave + 16u * (uint32_t)it;   // tile 0..255
    const uint32_t b = t * 16u + (uint32_t)n;                 // base 0..4095
    constexpr uint32_t lowmask = PNIB ? ((1u << (4 * PNIB)) - 1u) : 0u;
    const uint32_t idx0 = ((b >> (4 * PNIB)) << (4 * PNIB + 4)) | (b & lowmask);
    const uint32_t pb = lam<LAM>(idx0);                       // runtime bit-chain

    // B fragment: 16 K-values of this lane's half (Re or Im) of subcube `b`
    v16h bf;
#pragma unroll
    for (int k = 0; k < 16; ++k)
      bf[k] = src[pb ^ lam<LAM>((uint32_t)k << (4 * PNIB))];  // constant-folded offsets

    if (PRE) {   // attn1: |psi4|^2 with per-wire signs (logical index bits)
#pragma unroll
      for (int k = 0; k < 16; ++k) {
        float v = (float)bf[k]; float v2 = v * v;
        uint32_t li = idx0 | ((uint32_t)k << (4 * PNIB));
        ms.aTot += v2;
#pragma unroll
        for (int w = 0; w < 16; ++w)
          if ((li >> (15 - w)) & 1u) ms.aBit[w] += v2;
      }
    }

    v8f cz = {};
    v8f dT = __builtin_amdgcn_wmma_f32_16x16x32_f16(false, aT, false, bf, (short)0, cz, false, false);
    v8f dB = __builtin_amdgcn_wmma_f32_16x16x32_f16(false, aB, false, bf, (short)0, cz, false, false);

    if (POST) {  // m1: |psi5|^2, lane holds Re (dT) and Im (dB) of its 8 amps
#pragma unroll
      for (int r = 0; r < 8; ++r) {
        float v2 = dT[r] * dT[r] + dB[r] * dB[r];
        uint32_t li = idx0 | ((uint32_t)r << (4 * PNIB)) | (im ? (8u << (4 * PNIB)) : 0u);
        ms.mTot += v2;
#pragma unroll
        for (int w = 0; w < 16; ++w)
          if ((li >> (15 - w)) & 1u) ms.mBit[w] += v2;
      }
    }
    if (STORE) {
#pragma unroll
      for (int r = 0; r < 8; ++r) {
        uint32_t off = pb ^ lam<LAM>((uint32_t)r << (4 * PNIB)) ^ imoff;
        Re[off] = (_Float16)dT[r];
        Im[off] = (_Float16)dB[r];
      }
    }
  }
  __syncthreads();
}

// ---------------- setup: build six 32x32 f16 stage matrices from angles ----------
struct c2 { float x, y; };
__device__ __forceinline__ c2 cmul(c2 a, c2 b) { return {a.x * b.x - a.y * b.y, a.x * b.y + a.y * b.x}; }
__device__ __forceinline__ c2 cadd(c2 a, c2 b) { return {a.x + b.x, a.y + b.y}; }
struct m22 { c2 e[2][2]; };
__device__ m22 mmul(const m22& A, const m22& B) {
  m22 C;
  for (int i = 0; i < 2; ++i)
    for (int j = 0; j < 2; ++j)
      C.e[i][j] = cadd(cmul(A.e[i][0], B.e[0][j]), cmul(A.e[i][1], B.e[1][j]));
  return C;
}
__device__ m22 rxm(float th) {
  float c = cosf(0.5f * th), s = sinf(0.5f * th);
  m22 M; M.e[0][0] = {c, 0}; M.e[0][1] = {0, -s}; M.e[1][0] = {0, -s}; M.e[1][1] = {c, 0};
  return M;
}
__device__ m22 rym(float th) {
  float c = cosf(0.5f * th), s = sinf(0.5f * th);
  m22 M; M.e[0][0] = {c, 0}; M.e[0][1] = {-s, 0}; M.e[1][0] = {s, 0}; M.e[1][1] = {c, 0};
  return M;
}

__global__ void qsim_setup_kernel(const float* __restrict__ ra, const float* __restrict__ trx,
                                  const float* __restrict__ trY, _Float16* __restrict__ mats) {
  __shared__ c2 U[18][2][2];
  if (threadIdx.x == 0) {
    m22 RX0 = rxm(trx[0]);
    m22 W[18];
    W[0]  = mmul(RX0, mmul(rym(ra[16]), rxm(ra[0])));   // wire0: RX(a0),RY(a16) then blk0 RX
    W[1]  = mmul(RX0, rym(ra[13]));
    W[2]  = mmul(RX0, rym(ra[10]));
    W[3]  = mmul(RX0, rym(ra[7]));
    W[4]  = mmul(RX0, rym(ra[4]));
    W[5]  = RX0;                                        // RY(a1) applied during load pass
    W[6]  = RX0; W[7] = RX0; W[8] = RX0;
    W[9]  = mmul(RX0, rym(ra[19]));                     // post-perm rotations folded in
    W[10] = mmul(RX0, rxm(ra[18]));
    W[11] = mmul(RX0, rxm(ra[15]));
    W[12] = mmul(RX0, rxm(ra[12]));
    W[13] = mmul(RX0, rxm(ra[9]));
    W[14] = mmul(RX0, rxm(ra[6]));
    W[15] = mmul(RX0, rxm(ra[3]));
    W[16] = mmul(rxm(trx[1]), rym(trY[0]));             // merged blk0-RY + blk1-RX
    W[17] = rym(trY[1]);                                // blk1 RY
    for (int i = 0; i < 18; ++i)
      for (int r = 0; r < 2; ++r)
        for (int c = 0; c < 2; ++c) U[i][r][c] = W[i].e[r][c];
  }
  __syncthreads();
  // mats: [0..3]=stage C groups, [4]=stage E, [5]=stage R; each 32x32 real f16 row-major
  for (int e = threadIdx.x; e < 6 * 1024; e += blockDim.x) {
    int matid = e >> 10, rc = e & 1023, r32 = rc >> 5, c32 = rc & 31;
    int r = r32 & 15, c = c32 & 15;
    c2 p = {1.f, 0.f};
    for (int d = 0; d < 4; ++d) {
      int w = (matid < 4) ? (4 * matid + d) : (matid == 4 ? 16 : 17);
      p = cmul(p, U[w][(r >> (3 - d)) & 1][(c >> (3 - d)) & 1]);
    }
    bool rI = r32 >= 16, cI = c32 >= 16;
    float val = !rI ? (cI ? -p.y : p.x) : (cI ? p.x : p.y);   // [[Re,-Im],[Im,Re]]
    mats[e] = (_Float16)val;
  }
}

// ---------------- main: whole statevector resident in LDS (fp16 SoA, 256 KB) -----
__device__ __forceinline__ float wred(float v) {
  for (int o = 16; o >= 1; o >>= 1) v += __shfl_xor(v, o, 32);
  return v;
}

__global__ void __launch_bounds__(THREADS)
qsim_main_kernel(const float* __restrict__ states, const float* __restrict__ ra,
                 const _Float16* __restrict__ mats,
                 const float* __restrict__ w1p, const float* __restrict__ b1p,
                 const float* __restrict__ w2p, const float* __restrict__ b2p,
                 const float* __restrict__ g1p, const float* __restrict__ be1p,
                 const float* __restrict__ g2p, const float* __restrict__ be2p,
                 const float* __restrict__ whp, const float* __restrict__ bhp,
                 float* __restrict__ out) {
  extern __shared__ char smem[];
  _Float16* Re = (_Float16*)smem;
  _Float16* Im = Re + NSTATE;
  float* red = (float*)(Im + NSTATE);          // 64 floats reduction scratch
  const int batch = blockIdx.x;

  if (threadIdx.x < 64) red[threadIdx.x] = 0.f;

  // load pass: global f32 -> fp16 LDS, with RY(rand_angles[1]) on wire 5 fused in.
  {
    float a1 = ra[1];
    float c5 = cosf(0.5f * a1), s5 = sinf(0.5f * a1);
    const float* S = states + (size_t)batch * NSTATE;
    for (int pi = threadIdx.x; pi < NSTATE / 2; pi += THREADS) {
      uint32_t l0 = ((uint32_t)(pi >> 10) << 11) | ((uint32_t)pi & 1023u);  // bit10 = wire5
      uint32_t l1 = l0 | 1024u;
      float g0 = S[l0], g1v = S[l1];
      Re[l0] = (_Float16)(c5 * g0 - s5 * g1v);
      Re[l1] = (_Float16)(s5 * g0 + c5 * g1v);
      Im[l0] = (_Float16)0.f; Im[l1] = (_Float16)0.f;
    }
  }
  __syncthreads();

  Meas ms;
  ms.aTot = 0.f; ms.mTot = 0.f;
#pragma unroll
  for (int w = 0; w < 16; ++w) { ms.aBit[w] = 0.f; ms.mBit[w] = 0.f; }

  // stage C: per-wire [random-layer rotations] + blk0 RX, layout = after perm P
  stage_pass<1, 3, false, false, true>(Re, Im, mats + 0 * 1024, ms);
  stage_pass<1, 2, false, false, true>(Re, Im, mats + 1 * 1024, ms);
  stage_pass<1, 1, false, false, true>(Re, Im, mats + 2 * 1024, ms);
  stage_pass<1, 0, false, false, true>(Re, Im, mats + 3 * 1024, ms);
  // stage E: RX(tx1)*RY(ty0) on every wire, layout = after first ring
  stage_pass<2, 3, false, false, true>(Re, Im, mats + 4 * 1024, ms);
  stage_pass<2, 2, false, false, true>(Re, Im, mats + 4 * 1024, ms);
  stage_pass<2, 1, false, false, true>(Re, Im, mats + 4 * 1024, ms);
  stage_pass<2, 0, false, false, true>(Re, Im, mats + 4 * 1024, ms);
  // stage R: RY(ty1) on every wire, layout = after second ring.
  // First pass measures attn1 (pre), last pass measures m1 (post, no store needed).
  stage_pass<3, 3, true,  false, true >(Re, Im, mats + 5 * 1024, ms);
  stage_pass<3, 2, false, false, true >(Re, Im, mats + 5 * 1024, ms);
  stage_pass<3, 1, false, false, true >(Re, Im, mats + 5 * 1024, ms);
  stage_pass<3, 0, false, true,  false>(Re, Im, mats + 5 * 1024, ms);

  // reduce 34 accumulators: wave32 shuffle, then LDS atomics
  const int lane = threadIdx.x & 31;
  {
    float v = wred(ms.aTot); if (lane == 0) atomicAdd(&red[0], v);
#pragma unroll
    for (int w = 0; w < 16; ++w) { float r = wred(ms.aBit[w]); if (lane == 0) atomicAdd(&red[1 + w], r); }
    v = wred(ms.mTot); if (lane == 0) atomicAdd(&red[17], v);
#pragma unroll
    for (int w = 0; w < 16; ++w) { float r = wred(ms.mBit[w]); if (lane == 0) atomicAdd(&red[18 + w], r); }
  }
  __syncthreads();

  // tiny MLP head (block 1 parameters only), one lane per batch element
  if (threadIdx.x == 0) {
    float attn[16], m1[16];
    float aT = red[0], mT = red[17];
    for (int w = 0; w < 16; ++w) {
      attn[w] = (aT - 2.f * red[1 + w]) / aT;
      m1[w]   = (mT - 2.f * red[18 + w]) / mT;
    }
    // x = LN(attn; g1[1], be1[1])
    float mu = 0.f; for (int w = 0; w < 16; ++w) mu += attn[w]; mu *= (1.f / 16.f);
    float var = 0.f; for (int w = 0; w < 16; ++w) { float d = attn[w] - mu; var += d * d; } var *= (1.f / 16.f);
    float rs = rsqrtf(var + 1e-5f);
    float x[16];
    for (int w = 0; w < 16; ++w) x[w] = (attn[w] - mu) * rs * g1p[16 + w] + be1p[16 + w];
    // h = relu(m1 @ w1[1].T + b1[1]); ffn = h @ w2[1].T + b2[1]
    float h[64];
    for (int j = 0; j < 64; ++j) {
      float s = b1p[64 + j];
      for (int w = 0; w < 16; ++w) s += m1[w] * w1p[1024 + j * 16 + w];
      h[j] = fmaxf(s, 0.f);
    }
    float y[16];
    for (int w = 0; w < 16; ++w) {
      float s = b2p[16 + w];
      for (int j = 0; j < 64; ++j) s += w2p[1024 + w * 64 + j] * h[j];
      y[w] = x[w] + s;
    }
    // feats = LN(y; g2[1], be2[1]); out = feats @ wh.T + bh
    mu = 0.f; for (int w = 0; w < 16; ++w) mu += y[w]; mu *= (1.f / 16.f);
    var = 0.f; for (int w = 0; w < 16; ++w) { float d = y[w] - mu; var += d * d; } var *= (1.f / 16.f);
    rs = rsqrtf(var + 1e-5f);
    float o = bhp[0];
    for (int w = 0; w < 16; ++w) o += ((y[w] - mu) * rs * g2p[16 + w] + be2p[16 + w]) * whp[w];
    out[batch] = o;
  }
}

extern "C" void kernel_launch(void* const* d_in, const int* in_sizes, int n_in,
                              void* d_out, int out_size, void* d_ws, size_t ws_size,
                              hipStream_t stream) {
  const float* states = (const float*)d_in[0];
  const float* ra     = (const float*)d_in[1];
  const float* trx    = (const float*)d_in[2];
  const float* trY    = (const float*)d_in[3];
  const float* w1p    = (const float*)d_in[4];
  const float* b1p    = (const float*)d_in[5];
  const float* w2p    = (const float*)d_in[6];
  const float* b2p    = (const float*)d_in[7];
  const float* g1p    = (const float*)d_in[8];
  const float* be1p   = (const float*)d_in[9];
  const float* g2p    = (const float*)d_in[10];
  const float* be2p   = (const float*)d_in[11];
  const float* whp    = (const float*)d_in[12];
  const float* bhp    = (const float*)d_in[13];
  _Float16* mats = (_Float16*)d_ws;   // 6 * 1024 halves = 12 KB

  qsim_setup_kernel<<<1, 256, 0, stream>>>(ra, trx, trY, mats);

  size_t shmem = (size_t)2 * NSTATE * sizeof(_Float16) + 64 * sizeof(float); // ~262 KB of 320 KB
  (void)hipFuncSetAttribute(reinterpret_cast<const void*>(qsim_main_kernel),
                            hipFuncAttributeMaxDynamicSharedMemorySize, (int)shmem);
  qsim_main_kernel<<<BSZ, THREADS, shmem, stream>>>(
      states, ra, mats, w1p, b1p, w2p, b2p, g1p, be1p, g2p, be2p, whp, bhp, (float*)d_out);
}
